// CausalSelfAttention_9594956939421
// MI455X (gfx1250) — matrix operand order repaired and checked by reading, not compiled, not run on hardware
//
#include <hip/hip_runtime.h>
#include <hip/hip_bf16.h>

// ---------------------------------------------------------------------------
// CausalSelfAttention forward for MI455X (gfx1250, wave32, WMMA bf16 + TDM)
// B=2, T=2048, C=2048, H=16, Dh=128
// ---------------------------------------------------------------------------

typedef __bf16 bf16_t;
typedef __attribute__((ext_vector_type(16))) bf16_t v16bf;
typedef __attribute__((ext_vector_type(8)))  float  v8f;
typedef __attribute__((ext_vector_type(4)))  unsigned u32x4;
typedef __attribute__((ext_vector_type(8)))  int      i32x8;
typedef __attribute__((ext_vector_type(4)))  int      i32x4;

#define TDIM 2048
#define CDIM 2048
#define HNUM 16
#define DH   128
#define C3   6144
#define MROWS 4096   // B*T

// LDS strides (ushorts): multiples of 8 -> 16B-aligned rows -> ds_load_b128
// fragment fetches; stride-in-dwords shares gcd 4 with 64 banks -> the 16
// lanes of each half cover 16 distinct bank quads (conflict-free for b128).
#define XAP 72    // GEMM A tile [128][72]
#define WBP 72    // GEMM W tile [64][72]
#define KTP 136   // attn K tile [64][136]  (Dh=128 + 8 pad)
#define VTP 72    // attn V^T tile [128][72]
#define PLP 72    // attn P tile [16][72] per wave

__device__ __forceinline__ bf16_t us2bf(unsigned short u) {
    return __builtin_bit_cast(bf16_t, u);
}
__device__ __forceinline__ unsigned short f2bf(float x) {
    unsigned u = __builtin_bit_cast(unsigned, x);
    u += 0x7FFFu + ((u >> 16) & 1u);          // round-to-nearest-even
    return (unsigned short)(u >> 16);
}

// A-matrix (16x32, 16-bit) element->K map: lanes0-15 K{0..7,16..23}; +8 for hi half
__device__ __forceinline__ int kmapA(int e, int h) { return e + (h << 3) + (e & 8); }
// B-matrix (32x16, 16-bit) element->K map: same as A: lanes0-15 K{0..7,16..23}; +8 for hi half
__device__ __forceinline__ int kmapB(int e, int h) { return e + (h << 3) + (e & 8); }

__device__ __forceinline__ v16bf load_fragA(const unsigned short* p, int rowBase,
                                            int stride, int kBase, int lane) {
    v16bf f;
    int r = rowBase + (lane & 15);
    int h = (lane >> 4) & 1;
#pragma unroll
    for (int e = 0; e < 16; ++e)
        f[e] = us2bf(p[r * stride + kBase + kmapA(e, h)]);
    return f;
}
__device__ __forceinline__ v16bf load_fragB(const unsigned short* p, int nBase,
                                            int stride, int kBase, int lane) {
    v16bf f;
    int n = nBase + (lane & 15);
    int h = (lane >> 4) & 1;
#pragma unroll
    for (int e = 0; e < 16; ++e)
        f[e] = us2bf(p[n * stride + kBase + kmapB(e, h)]);
    return f;
}

__device__ __forceinline__ v8f wmma_bf16(v16bf a, v16bf b, v8f c) {
    return __builtin_amdgcn_wmma_f32_16x16x32_bf16(false, a, false, b,
                                                   (short)0, c, false, false);
}

__device__ __forceinline__ float rmax16(float v) {
    v = fmaxf(v, __shfl_xor(v, 1, 32));
    v = fmaxf(v, __shfl_xor(v, 2, 32));
    v = fmaxf(v, __shfl_xor(v, 4, 32));
    v = fmaxf(v, __shfl_xor(v, 8, 32));
    return v;
}
__device__ __forceinline__ float rsum16(float v) {
    v += __shfl_xor(v, 1, 32);
    v += __shfl_xor(v, 2, 32);
    v += __shfl_xor(v, 4, 32);
    v += __shfl_xor(v, 8, 32);
    return v;
}

// ---------------------------------------------------------------------------
// Tensor Data Mover: 2-D bf16 tile Global -> LDS, with LDS row padding.
// D# per CDNA5 ISA ch.8: group0 = {count, lds_addr, global_addr, type=2},
// group1 = {data_size=2B, pad cfg, tensor dims, tile dims, dim0 stride}.
// ---------------------------------------------------------------------------
__device__ __forceinline__ void tdm_load_tile_2d(unsigned lds_byte_addr,
                                                 const void* gptr,
                                                 unsigned tensor_d0, unsigned tensor_d1,
                                                 unsigned tile_d0, unsigned tile_d1,
                                                 unsigned stride0_elems,
                                                 int pad_interval_code,
                                                 int pad_amount_code) {
#if defined(__HIP_DEVICE_COMPILE__)
    unsigned long long ga = (unsigned long long)(size_t)gptr;
    u32x4 g0;
    g0[0] = 1u;                                                  // count=1
    g0[1] = lds_byte_addr;                                       // lds_addr
    g0[2] = (unsigned)(ga & 0xFFFFFFFFull);                      // global[31:0]
    g0[3] = (unsigned)((ga >> 32) & 0x1FFFFFFull) | (2u << 30);  // global[56:32]+type=2
    i32x8 g1;
    g1[0] = (1 << 16) | (1 << 20) |                       // data_size=2B, pad_enable
            (pad_interval_code << 22) | (pad_amount_code << 25);
    g1[1] = (int)((tensor_d0 & 0xFFFFu) << 16);           // tensor_dim0[15:0]
    g1[2] = (int)((tensor_d0 >> 16) | ((tensor_d1 & 0xFFFFu) << 16));
    g1[3] = (int)((tensor_d1 >> 16) | (tile_d0 << 16));   // tile_dim0
    g1[4] = (int)(tile_d1 & 0xFFFFu);                     // tile_dim1 (tile_dim2=0)
    g1[5] = (int)stride0_elems;                           // tensor_dim0_stride[31:0]
    g1[6] = 0;
    g1[7] = 0;
    i32x4 z4 = {0, 0, 0, 0};
#if __clang_major__ >= 23
    i32x8 z8 = {0, 0, 0, 0, 0, 0, 0, 0};
    __builtin_amdgcn_tensor_load_to_lds(g0, g1, z4, z4, z8, 0);
#else
    __builtin_amdgcn_tensor_load_to_lds(g0, g1, z4, z4, 0);
#endif
#endif
}

// ---------------------------------------------------------------------------
// GEMM: 128x64 tile per 128-thread block; each wave owns 32 rows x 64 cols,
// K staged 64-deep (2 WMMA slabs) -> 16 WMMA per 12 fragment loads per stage.
// MODE 0: A = x fp32, W = w_qkv (ldN=6144); epilogue RoPE + q/k/v scatter.
// MODE 1: A = y bf16 ws, W = w_proj (ldN=2048); epilogue fp32 store.
// ---------------------------------------------------------------------------
template <int MODE>
__global__ void __launch_bounds__(128)
gemm128x64_kernel(const float* __restrict__ Af32,
                  const unsigned short* __restrict__ Abf16,
                  const float* __restrict__ Wg, int ldN,
                  float* __restrict__ out,
                  unsigned short* __restrict__ qh,
                  unsigned short* __restrict__ kh,
                  unsigned short* __restrict__ vh) {
    __shared__ __align__(16) unsigned short xl[128 * XAP];  // A tile [M=128][K=64]
    __shared__ __align__(16) unsigned short wl[64 * WBP];   // W tile [N=64][K=64]

    const int tid  = threadIdx.x;
    const int lane = tid & 31;
    const int wave = tid >> 5;
    const int half = (lane >> 4) & 1;
    const int ln   = lane & 15;

    const int n0 = blockIdx.x * 64;
    const int m0 = blockIdx.y * 128;

    v8f acc[2][4];
    const v8f vzero = {0.f, 0.f, 0.f, 0.f, 0.f, 0.f, 0.f, 0.f};
#pragma unroll
    for (int g = 0; g < 2; ++g)
#pragma unroll
        for (int s = 0; s < 4; ++s) acc[g][s] = vzero;

    for (int k0 = 0; k0 < CDIM; k0 += 64) {
        __syncthreads();
        // ---- stage A tile: [128][64] -> LDS [row][k] ----
        if (MODE == 0) {
#pragma unroll
            for (int j = 0; j < 16; ++j) {
                int flat = (tid + j * 128) * 4;       // 128*64 floats
                int row  = flat >> 6;
                int k    = flat & 63;
                float4 xv = *(const float4*)(Af32 + (m0 + row) * CDIM + k0 + k);
                xl[row * XAP + k + 0] = f2bf(xv.x);
                xl[row * XAP + k + 1] = f2bf(xv.y);
                xl[row * XAP + k + 2] = f2bf(xv.z);
                xl[row * XAP + k + 3] = f2bf(xv.w);
            }
        } else {
#pragma unroll
            for (int j = 0; j < 8; ++j) {
                int flat = (tid + j * 128) * 8;       // 128*64 ushorts
                int row  = flat >> 6;
                int k    = flat & 63;
                uint4 av = *(const uint4*)(Abf16 + (m0 + row) * CDIM + k0 + k);
                unsigned short ab[8];
                __builtin_memcpy(ab, &av, 16);
#pragma unroll
                for (int q = 0; q < 8; ++q) xl[row * XAP + k + q] = ab[q];
            }
        }
        // ---- stage W tile transposed: global [K=64][N=64] -> LDS [n][k] ----
#pragma unroll
        for (int j = 0; j < 8; ++j) {
            int flat = (tid + j * 128) * 4;           // 64*64 floats
            int kk   = flat >> 6;
            int n    = flat & 63;
            float4 wv = *(const float4*)(Wg + (size_t)(k0 + kk) * ldN + n0 + n);
            wl[(n + 0) * WBP + kk] = f2bf(wv.x);
            wl[(n + 1) * WBP + kk] = f2bf(wv.y);
            wl[(n + 2) * WBP + kk] = f2bf(wv.z);
            wl[(n + 3) * WBP + kk] = f2bf(wv.w);
        }
        if (k0 + 64 < CDIM)
            __builtin_prefetch(Wg + (size_t)(k0 + 64) * ldN + n0 + (tid & 63), 0, 1);
        __syncthreads();

        // ---- 16 WMMAs per wave ----
#pragma unroll
        for (int ks = 0; ks < 2; ++ks) {
            int kb = ks * 32;
            v16bf a0 = load_fragA(xl, wave * 32,      XAP, kb, lane);
            v16bf a1 = load_fragA(xl, wave * 32 + 16, XAP, kb, lane);
#pragma unroll
            for (int sub = 0; sub < 4; ++sub) {
                v16bf b = load_fragB(wl, sub * 16, WBP, kb, lane);
                acc[0][sub] = wmma_bf16(a0, b, acc[0][sub]);
                acc[1][sub] = wmma_bf16(a1, b, acc[1][sub]);
            }
        }
    }

    // ---- epilogue ----
    if (MODE == 1) {
#pragma unroll
        for (int g = 0; g < 2; ++g)
#pragma unroll
            for (int sub = 0; sub < 4; ++sub)
#pragma unroll
                for (int r = 0; r < 8; ++r) {
                    int m = m0 + wave * 32 + g * 16 + r + 8 * half;
                    out[(size_t)m * CDIM + n0 + sub * 16 + ln] = acc[g][sub][r];
                }
    } else {
#pragma unroll
        for (int g = 0; g < 2; ++g)
#pragma unroll
            for (int sub = 0; sub < 4; ++sub) {
                int n = n0 + sub * 16 + ln;      // q/k/v side uniform per wave
#pragma unroll
                for (int r = 0; r < 8; ++r) {
                    int m  = m0 + wave * 32 + g * 16 + r + 8 * half;
                    int t  = m & (TDIM - 1);
                    int bb = m >> 11;
                    float val = acc[g][sub][r];
                    if (n < 2 * CDIM) {          // q or k: RoPE
                        float part = __shfl_xor(val, 1, 32);  // pair partner d^1
                        int   c    = n & (CDIM - 1);
                        int   hh   = c >> 7;
                        int   d    = c & 127;
                        float freq = __powf(10000.f, -(float)(d & ~1) * (1.0f / 128.0f));
                        float ang  = (float)t * freq;
                        float cs   = __cosf(ang), sn = __sinf(ang);
                        float o    = (d & 1) ? (part * sn + val * cs)
                                             : (val * cs - part * sn);
                        size_t addr = ((size_t)(bb * HNUM + hh) * TDIM + t) * DH + d;
                        if (n < CDIM) qh[addr] = f2bf(o);
                        else          kh[addr] = f2bf(o);
                    } else {                     // v: straight copy
                        int   c  = n - 2 * CDIM;
                        int   hh = c >> 7;
                        int   d  = c & 127;
                        size_t addr = ((size_t)(bb * HNUM + hh) * TDIM + t) * DH + d;
                        vh[addr] = f2bf(val);
                    }
                }
            }
    }
}

// ---------------------------------------------------------------------------
// Flash attention: block = one (b,h) x 64 query rows; 4 waves x 16 rows.
// K tile staged by the Tensor Data Mover (TDM pad cfg gives the 136 stride);
// V tile staged transposed manually (TDM cannot transpose); Q.K^T and P.V on
// WMMA bf16 with online softmax.
// ---------------------------------------------------------------------------
__global__ void __launch_bounds__(128)
attn_kernel(const unsigned short* __restrict__ qh,
            const unsigned short* __restrict__ kh,
            const unsigned short* __restrict__ vh,
            unsigned short* __restrict__ yh) {
    __shared__ __align__(16) unsigned short Ktl[64 * KTP];   // [key][d]
    __shared__ __align__(16) unsigned short Vtl[128 * VTP];  // [d][key]
    __shared__ __align__(16) unsigned short Pl[4 * 16 * PLP];

    const int tid  = threadIdx.x;
    const int lane = tid & 31;
    const int wave = tid >> 5;
    const int half = (lane >> 4) & 1;
    const int ln   = lane & 15;

    const int qt = blockIdx.x;
    const int bh = blockIdx.y;
    const int base = bh * (TDIM * DH);
    const int t0 = qt * 64 + wave * 16;

    // ---- Q fragments (A layout), 16 rows x 128 d ----
    v16bf qf[4];
#pragma unroll
    for (int c = 0; c < 4; ++c)
#pragma unroll
        for (int e = 0; e < 16; ++e) {
            int d = c * 32 + kmapA(e, half);
            qf[c][e] = us2bf(qh[base + (t0 + ln) * DH + d]);
        }

    float mrow[8], lrow[8];
    v8f yacc[8];
    const v8f vzero = {0.f, 0.f, 0.f, 0.f, 0.f, 0.f, 0.f, 0.f};
#pragma unroll
    for (int r = 0; r < 8; ++r) { mrow[r] = -1e30f; lrow[r] = 0.f; }
#pragma unroll
    for (int nc = 0; nc < 8; ++nc) yacc[nc] = vzero;

    const float scale = 0.08838834764831845f;   // 1/sqrt(128)
    const unsigned ldsK = (unsigned)(size_t)(const void*)Ktl;

    for (int kt = 0; kt <= qt; ++kt) {
        __syncthreads();
        const unsigned short* kp = kh + base + kt * 64 * DH;
        const unsigned short* vp = vh + base + kt * 64 * DH;

        // K tile: TDM async DMA, one descriptor from wave 0.
        // pad_interval code 5 = 64 dwords (one 128-elem bf16 row),
        // pad_amount  code 3 = 4 dwords -> LDS row stride 136 ushorts.
        if (wave == 0)
            tdm_load_tile_2d(ldsK, kp, DH, TDIM, DH, 64, DH, 5, 3);

        // V tile: manual transposed staging [d][key] (overlaps the TDM).
#pragma unroll
        for (int j = 0; j < 8; ++j) {
            int flat = (tid + j * 128) * 8;     // 64*128 ushorts
            int key  = flat >> 7;
            int d    = flat & 127;
            uint4 vv = *(const uint4*)(vp + key * DH + d);
            unsigned short vb[8];
            __builtin_memcpy(vb, &vv, 16);
#pragma unroll
            for (int q = 0; q < 8; ++q) Vtl[(d + q) * VTP + key] = vb[q];
        }
        if (wave == 0) __builtin_amdgcn_s_wait_tensorcnt(0);
        __syncthreads();

        // ---- S = Q.K^T (16 x 64 per wave, K-depth 128) ----
        v8f s[4];
#pragma unroll
        for (int sub = 0; sub < 4; ++sub) s[sub] = vzero;
#pragma unroll
        for (int c = 0; c < 4; ++c) {
            v16bf a = qf[c];
#pragma unroll
            for (int sub = 0; sub < 4; ++sub) {
                v16bf b = load_fragB(Ktl, sub * 16, KTP, c * 32, lane);
                s[sub] = wmma_bf16(a, b, s[sub]);
            }
        }

        // ---- causal mask + online softmax ----
#pragma unroll
        for (int r = 0; r < 8; ++r) {
            const int i = t0 + r + 8 * half;
            float mx = -1e30f;
#pragma unroll
            for (int sub = 0; sub < 4; ++sub) {
                int j = kt * 64 + sub * 16 + ln;
                float v = s[sub][r] * scale;
                v = (j <= i) ? v : -1e30f;
                s[sub][r] = v;
                mx = fmaxf(mx, v);
            }
            mx = rmax16(mx);
            float mnew  = fmaxf(mrow[r], mx);
            float alpha = __expf(mrow[r] - mnew);
            float rs = 0.f;
#pragma unroll
            for (int sub = 0; sub < 4; ++sub) {
                float p = __expf(s[sub][r] - mnew);
                Pl[(wave * 16 + r + 8 * half) * PLP + sub * 16 + ln] = f2bf(p);
                rs += p;
            }
            rs = rsum16(rs);
            lrow[r] = lrow[r] * alpha + rs;
            mrow[r] = mnew;
#pragma unroll
            for (int nc = 0; nc < 8; ++nc) yacc[nc][r] *= alpha;
        }
        __syncthreads();

        // ---- Y += P.V (16 x 128 per wave, K-depth 64) ----
#pragma unroll
        for (int c = 0; c < 2; ++c) {
            v16bf a = load_fragA(Pl + wave * 16 * PLP, 0, PLP, c * 32, lane);
#pragma unroll
            for (int nc = 0; nc < 8; ++nc) {
                v16bf b = load_fragB(Vtl, nc * 16, VTP, c * 32, lane);
                yacc[nc] = wmma_bf16(a, b, yacc[nc]);
            }
        }
    }

    // ---- normalize and store y (bf16 [B,T,C], C = h*128 + d) ----
    const int bb = bh >> 4;
    const int hh = bh & 15;
#pragma unroll
    for (int r = 0; r < 8; ++r) {
        float inv = 1.0f / lrow[r];
        int t = t0 + r + 8 * half;
#pragma unroll
        for (int nc = 0; nc < 8; ++nc) {
            int ch = hh * DH + nc * 16 + ln;
            yh[(size_t)(bb * TDIM + t) * CDIM + ch] = f2bf(yacc[nc][r] * inv);
        }
    }
}

// ---------------------------------------------------------------------------
extern "C" void kernel_launch(void* const* d_in, const int* in_sizes, int n_in,
                              void* d_out, int out_size, void* d_ws, size_t ws_size,
                              hipStream_t stream) {
    const float* x      = (const float*)d_in[0];   // [2,2048,2048]
    const float* w_qkv  = (const float*)d_in[1];   // [2048,6144]
    const float* w_proj = (const float*)d_in[2];   // [2048,2048]
    float* out = (float*)d_out;                    // [2,2048,2048]

    const size_t QE = (size_t)2 * HNUM * TDIM * DH;  // 8,388,608 bf16 elems
    unsigned short* qh = (unsigned short*)d_ws;
    unsigned short* kh = qh + QE;
    unsigned short* vh = kh + QE;
    unsigned short* yh = vh + QE;

    // 1) qkv = x @ w_qkv, fused RoPE, scatter to [B,H,T,Dh] bf16
    gemm128x64_kernel<0><<<dim3(C3 / 64, MROWS / 128), 128, 0, stream>>>(
        x, nullptr, w_qkv, C3, nullptr, qh, kh, vh);

    // 2) flash attention -> y bf16 [B,T,C]
    attn_kernel<<<dim3(TDIM / 64, 2 * HNUM), 128, 0, stream>>>(qh, kh, vh, yh);

    // 3) out = y @ w_proj (fp32 out)
    gemm128x64_kernel<1><<<dim3(CDIM / 64, MROWS / 128), 128, 0, stream>>>(
        nullptr, yh, w_proj, CDIM, out, nullptr, nullptr, nullptr);
}
